// VirtualTwinAttentionLayer_11373073399941
// MI455X (gfx1250) — compile-verified, matrix operand order checked by reading, not run on hardware
//
#include <hip/hip_runtime.h>
#include <hip/hip_bf16.h>

typedef _Float16 half_t;
typedef __attribute__((ext_vector_type(16))) _Float16 v16h;
typedef __attribute__((ext_vector_type(8)))  float    v8f;

#define NN     512
#define INF_   128
#define OUTF   64
#define NH     4
#define HF     256   // NH * OUTF
#define NEGBIG (-9000000000000000.0f)

// ---------------- WMMA fragment loaders (wave32, 16x16x32 f16) ----------------
// A (16xK tile) or B^T stored row-major [rows, ld] with K along the row.
// lane 0-15: row = lane, K = kb + {0..7, 16..23}
// lane 16-31: row = lane-16, K = kb + {8..15, 24..31}
__device__ __forceinline__ v16h load_frag_nk(const half_t* __restrict__ p, int ld,
                                             int rowbase, int kbase, int lane) {
    int r    = lane & 15;
    int koff = (lane >> 4) << 3;
    const half_t* row = p + (size_t)(rowbase + r) * ld + kbase;
    v16h v;
#pragma unroll
    for (int e = 0; e < 8; ++e) v[e] = row[koff + e];
#pragma unroll
    for (int e = 0; e < 8; ++e) v[8 + e] = row[16 + koff + e];
    return v;
}

// B stored [K, N] row-major (ld = row stride): lane%16 selects column n.
__device__ __forceinline__ v16h load_frag_kn(const half_t* __restrict__ p, int ld,
                                             int kbase, int nbase, int lane) {
    int c    = lane & 15;
    int koff = (lane >> 4) << 3;
    v16h v;
#pragma unroll
    for (int e = 0; e < 8; ++e) v[e] = p[(size_t)(kbase + koff + e) * ld + nbase + c];
#pragma unroll
    for (int e = 0; e < 8; ++e) v[8 + e] = p[(size_t)(kbase + 16 + koff + e) * ld + nbase + c];
    return v;
}

#define WMMA_F16(a, b, c) \
    __builtin_amdgcn_wmma_f32_16x16x32_f16(false, (a), false, (b), (short)0, (c), false, false)

// ---------------- kernels ----------------
__global__ void cast_f32_f16(const float* __restrict__ s, half_t* __restrict__ d, int n) {
    int t = blockIdx.x * blockDim.x + threadIdx.x;
    if (t < n) d[t] = (half_t)s[t];
}

// C = A[M,K] @ B[N,K]^T ; writes f32 and f16 copies (either may be null)
__global__ void gemm_abt(const half_t* __restrict__ A, const half_t* __restrict__ B,
                         float* __restrict__ Cf, half_t* __restrict__ Ch,
                         int M, int N, int K) {
    int wave = (blockIdx.x * blockDim.x + threadIdx.x) >> 5;
    int lane = threadIdx.x & 31;
    int ntiles = N >> 4;
    if (wave >= (M >> 4) * ntiles) return;
    int mt = wave / ntiles, nt = wave % ntiles;
    v8f acc = {};
    for (int kb = 0; kb < K; kb += 32) {
        __builtin_prefetch(A + (size_t)(mt * 16 + (lane & 15)) * K + kb + 32, 0, 0);
        v16h aa = load_frag_nk(A, K, mt * 16, kb, lane);
        v16h bb = load_frag_nk(B, K, nt * 16, kb, lane);
        acc = WMMA_F16(aa, bb, acc);
    }
    int n = lane & 15, mo = (lane >> 4) << 3;
#pragma unroll
    for (int r = 0; r < 8; ++r) {
        int m = mt * 16 + mo + r, c = nt * 16 + n;
        float v = acc[r];
        if (Cf) Cf[(size_t)m * N + c] = v;
        if (Ch) Ch[(size_t)m * N + c] = (half_t)v;
    }
}

// left[i,h,g] = sum_f ht[i, g*64+f] * a[h, f]     (t = i*16 + h*4 + g)
// right[j,h]  = sum_f ht[j, h*64+f] * a[h, 64+f]  (u = j*4 + h)
__global__ void scores_lr(const float* __restrict__ ht, const float* __restrict__ a,
                          float* __restrict__ left, float* __restrict__ right) {
    int t = blockIdx.x * blockDim.x + threadIdx.x;
    if (t < NN * NH * NH) {
        int i = t >> 4, h = (t >> 2) & 3, g = t & 3;
        float s = 0.f;
        for (int f = 0; f < OUTF; ++f)
            s += ht[(size_t)i * HF + g * OUTF + f] * a[h * 2 * OUTF + f];
        left[t] = s;
    } else {
        int u = t - NN * NH * NH;
        if (u < NN * NH) {
            int j = u >> 2, h = u & 3;
            float s = 0.f;
            for (int f = 0; f < OUTF; ++f)
                s += ht[(size_t)j * HF + h * OUTF + f] * a[h * 2 * OUTF + OUTF + f];
            right[u] = s;
        }
    }
}

// one block per row i (512 threads = one j each); softmax over j per head.
// alpha stored head-major: alpha[(h*NN + i)*NN + j]
__global__ void attn_softmax(const float* __restrict__ left, const float* __restrict__ right,
                             const float* __restrict__ ew, const int* __restrict__ adj,
                             half_t* __restrict__ alpha) {
    __shared__ float red[NN];
    int i = blockIdx.x;
    int j = threadIdx.x;
    int g = j >> 7;                      // the head-scramble from tile+reshape
    float ewv = ew[(size_t)i * NN + j];
    int   ad  = adj[(size_t)i * NN + j];
#pragma unroll
    for (int h = 0; h < NH; ++h) {
        float x = left[i * 16 + h * 4 + g] + right[j * 4 + h];
        x = (x > 0.f) ? x : 0.2f * x;    // leaky relu
        x *= ewv;
        x = (ad > 0) ? x : NEGBIG;
        red[j] = x; __syncthreads();
        for (int s = NN / 2; s > 0; s >>= 1) {
            if (j < s) red[j] = fmaxf(red[j], red[j + s]);
            __syncthreads();
        }
        float mx = red[0]; __syncthreads();
        float p = expf(x - mx);
        red[j] = p; __syncthreads();
        for (int s = NN / 2; s > 0; s >>= 1) {
            if (j < s) red[j] += red[j + s];
            __syncthreads();
        }
        float sum = red[0]; __syncthreads();
        alpha[((size_t)h * NN + i) * NN + j] = (half_t)(p / sum);
    }
}

// c[i, h*64+f] = sum_j alpha_h[i,j] * ht[j, h*64+f]  (per-head 512x512 @ 512x64)
__global__ void agg_gemm(const half_t* __restrict__ alpha, const half_t* __restrict__ ht16,
                         half_t* __restrict__ c16) {
    int wave = (blockIdx.x * blockDim.x + threadIdx.x) >> 5;
    int lane = threadIdx.x & 31;
    if (wave >= NH * (NN >> 4) * (OUTF >> 4)) return;
    int head = wave >> 7;
    int rem  = wave & 127;
    int mt = rem >> 2, nt = rem & 3;
    const half_t* A = alpha + (size_t)head * NN * NN;
    int nbase = head * OUTF + nt * 16;
    v8f acc = {};
    for (int kb = 0; kb < NN; kb += 32) {
        __builtin_prefetch(A + (size_t)(mt * 16 + (lane & 15)) * NN + kb + 32, 0, 0);
        v16h aa = load_frag_nk(A, NN, mt * 16, kb, lane);
        v16h bb = load_frag_kn(ht16, HF, kb, nbase, lane);
        acc = WMMA_F16(aa, bb, acc);
    }
    int n = lane & 15, mo = (lane >> 4) << 3;
#pragma unroll
    for (int r = 0; r < 8; ++r)
        c16[(size_t)(mt * 16 + mo + r) * HF + nbase + n] = (half_t)acc[r];
}

// out = sigmoid(h @ U^T + c @ V^T), [512, 256] f32
__global__ void final_gemm(const half_t* __restrict__ h16, const half_t* __restrict__ U16,
                           const half_t* __restrict__ c16, const half_t* __restrict__ V16,
                           float* __restrict__ out) {
    int wave = (blockIdx.x * blockDim.x + threadIdx.x) >> 5;
    int lane = threadIdx.x & 31;
    if (wave >= (NN >> 4) * (HF >> 4)) return;
    int mt = wave / (HF >> 4), nt = wave % (HF >> 4);
    v8f acc = {};
    for (int kb = 0; kb < INF_; kb += 32) {
        v16h aa = load_frag_nk(h16, INF_, mt * 16, kb, lane);
        v16h bb = load_frag_nk(U16, INF_, nt * 16, kb, lane);
        acc = WMMA_F16(aa, bb, acc);
    }
    for (int kb = 0; kb < HF; kb += 32) {
        v16h aa = load_frag_nk(c16, HF, mt * 16, kb, lane);
        v16h bb = load_frag_nk(V16, HF, nt * 16, kb, lane);
        acc = WMMA_F16(aa, bb, acc);
    }
    int n = lane & 15, mo = (lane >> 4) << 3;
#pragma unroll
    for (int r = 0; r < 8; ++r) {
        float v = acc[r];
        out[(size_t)(mt * 16 + mo + r) * HF + nt * 16 + n] = 1.0f / (1.0f + expf(-v));
    }
}

// ---------------- host ----------------
extern "C" void kernel_launch(void* const* d_in, const int* in_sizes, int n_in,
                              void* d_out, int out_size, void* d_ws, size_t ws_size,
                              hipStream_t stream) {
    const float* h   = (const float*)d_in[0];
    const int*   adj = (const int*)  d_in[1];
    const float* ew  = (const float*)d_in[2];
    const float* W   = (const float*)d_in[3];
    const float* a   = (const float*)d_in[4];
    const float* U   = (const float*)d_in[5];
    const float* V   = (const float*)d_in[6];
    float* out = (float*)d_out;

    char* ws = (char*)d_ws;
    size_t off = 0;
    auto alloc = [&](size_t bytes) -> void* {
        void* p = ws + off;
        off = (off + bytes + 255) & ~(size_t)255;
        return p;
    };
    half_t* h16   = (half_t*)alloc((size_t)NN * INF_ * 2);
    half_t* W16   = (half_t*)alloc((size_t)HF * INF_ * 2);
    half_t* U16   = (half_t*)alloc((size_t)HF * INF_ * 2);
    half_t* V16   = (half_t*)alloc((size_t)HF * HF * 2);
    float*  ht    = (float*) alloc((size_t)NN * HF * 4);
    half_t* ht16  = (half_t*)alloc((size_t)NN * HF * 2);
    float*  left  = (float*) alloc((size_t)NN * NH * NH * 4);
    float*  right = (float*) alloc((size_t)NN * NH * 4);
    half_t* alpha = (half_t*)alloc((size_t)NH * NN * NN * 2);
    half_t* c16   = (half_t*)alloc((size_t)NN * HF * 2);

    // f32 -> f16 casts
    cast_f32_f16<<<(NN * INF_ + 255) / 256, 256, 0, stream>>>(h, h16, NN * INF_);
    cast_f32_f16<<<(HF * INF_ + 255) / 256, 256, 0, stream>>>(W, W16, HF * INF_);
    cast_f32_f16<<<(HF * INF_ + 255) / 256, 256, 0, stream>>>(U, U16, HF * INF_);
    cast_f32_f16<<<(HF * HF + 255) / 256, 256, 0, stream>>>(V, V16, HF * HF);

    // ht = h @ W^T   (32 x 16 tiles = 512 waves; 4 waves/block)
    gemm_abt<<<128, 128, 0, stream>>>(h16, W16, ht, ht16, NN, HF, INF_);

    // left / right projections (8192 + 2048 threads)
    scores_lr<<<(NN * NH * NH + NN * NH + 255) / 256, 256, 0, stream>>>(ht, a, left, right);

    // masked softmax over neighbors, per (i, head)
    attn_softmax<<<NN, NN, 0, stream>>>(left, right, ew, adj, alpha);

    // c_h = alpha_h @ ht_h   (4 heads x 128 tiles = 512 waves)
    agg_gemm<<<128, 128, 0, stream>>>(alpha, ht16, c16);

    // out = sigmoid(h @ U^T + c @ V^T)
    final_gemm<<<128, 128, 0, stream>>>(h16, U16, c16, V16, out);
}
